// QuantizationLayer_49100066128350
// MI455X (gfx1250) — matrix operand order repaired and checked by reading, not compiled, not run on hardware
//
#include <hip/hip_runtime.h>
#include <stdint.h>

#define NN 32768
#define DD 512
#define KK 8192
#define D4 (DD / 4)          // 128 float4 per row
#define DP 520               // padded LDS row stride in elements (1040B: 16B-aligned, 4-bank skew)
#define ROWS 64              // rows per workgroup (2 row groups x 32)
#define NC 64                // codes per staged chunk (2 code groups x 32)
#define NCHUNK (KK / NC)     // 128
#define TPB 128              // 4 waves: (rowGrp 0/1) x (codeGrp 0/1), each 32 rows x 32 codes

typedef __attribute__((ext_vector_type(16))) __bf16 v16bf;
typedef __attribute__((ext_vector_type(8)))  float  v8f;
typedef unsigned int tdm_v4u __attribute__((ext_vector_type(4)));
typedef int          tdm_v8i __attribute__((ext_vector_type(8)));
typedef int          tdm_v4i __attribute__((ext_vector_type(4)));

union FragAB { uint4 u[2]; v16bf v; };

__device__ __forceinline__ unsigned short f32_to_bf16_rne(float f) {
    union { float f; uint32_t u; } c; c.f = f;
    return (unsigned short)((c.u + 0x7FFFu + ((c.u >> 16) & 1u)) >> 16);
}
__device__ __forceinline__ float bf16u_to_f32(unsigned short h) {
    union { uint32_t u; float f; } c; c.u = ((uint32_t)h) << 16;
    return c.f;
}

// ---- kernel: codebook squared norms (exact f32) ----------------------------
__global__ void vq_cnorm_kernel(const float* __restrict__ cb, float* __restrict__ cnorm) {
    int k = blockIdx.x * blockDim.x + threadIdx.x;
    if (k >= KK) return;
    const float4* row = (const float4*)(cb + (size_t)k * DD);
    float s = 0.f;
    #pragma unroll 8
    for (int i = 0; i < D4; ++i) {
        float4 v = row[i];
        s += v.x * v.x + v.y * v.y + v.z * v.z + v.w * v.w;
    }
    cnorm[k] = s;
}

// ---- kernel: pre-split codebook into bf16 hi/lo (row-major, once) ----------
__global__ void vq_split_cb_kernel(const float* __restrict__ cb,
                                   unsigned short* __restrict__ hi,
                                   unsigned short* __restrict__ lo) {
    size_t i = (size_t)blockIdx.x * blockDim.x + threadIdx.x;   // one float4
    float4 v = ((const float4*)cb)[i];
    float vv[4] = {v.x, v.y, v.z, v.w};
    unsigned short h[4], l[4];
    #pragma unroll
    for (int j = 0; j < 4; ++j) {
        h[j] = f32_to_bf16_rne(vv[j]);
        l[j] = f32_to_bf16_rne(vv[j] - bf16u_to_f32(h[j]));
    }
    ((ushort4*)hi)[i] = make_ushort4(h[0], h[1], h[2], h[3]);
    ((ushort4*)lo)[i] = make_ushort4(l[0], l[1], l[2], l[3]);
}

// ---- main kernel: bf16x3 WMMA distance GEMM + running argmin + gather ------
template <bool FROM_WS>
__launch_bounds__(TPB) __global__
void vq_main_kernel(const float* __restrict__ xg, const float* __restrict__ cb,
                    const unsigned short* __restrict__ cbhi,
                    const unsigned short* __restrict__ cblo,
                    const float* __restrict__ cnorm,
                    float* __restrict__ outq, float* __restrict__ outst) {
    __shared__ __align__(16) unsigned short xhi[ROWS * DP];
    __shared__ __align__(16) unsigned short xlo[ROWS * DP];
    __shared__ __align__(16) unsigned short chi[NC * DP];
    __shared__ __align__(16) unsigned short clo[NC * DP];
    __shared__ float svalm[2][ROWS];
    __shared__ int   sidxm[2][ROWS];
    __shared__ int   rowidx[ROWS];

    const int tid  = threadIdx.x;
    const int lane = tid & 31;
    const int wave = tid >> 5;
    const int r    = lane & 15;
    const int h    = lane >> 4;
    const int R0   = blockIdx.x * ROWS;
    const int rBase  = (wave & 1) * 32;   // this wave's 32 rows
    const int cgBase = (wave >> 1) * 32;  // this wave's 32 codes within chunk

    const float4* xg4 = (const float4*)xg;
    const float4* cb4 = (const float4*)cb;

    // ---- stage x tile once, split into bf16 hi/lo (padded layout) ----------
    for (int e = tid; e < ROWS * D4; e += TPB) {
        int row = e >> 7;
        int c   = e & (D4 - 1);
        float4 v = xg4[(size_t)(R0 + row) * D4 + c];
        float vv[4] = {v.x, v.y, v.z, v.w};
        unsigned short hi[4], lo[4];
        #pragma unroll
        for (int j = 0; j < 4; ++j) {
            hi[j] = f32_to_bf16_rne(vv[j]);
            lo[j] = f32_to_bf16_rne(vv[j] - bf16u_to_f32(hi[j]));
        }
        int o = row * DP + c * 4;
        *(ushort4*)(&xhi[o]) = make_ushort4(hi[0], hi[1], hi[2], hi[3]);
        *(ushort4*)(&xlo[o]) = make_ushort4(lo[0], lo[1], lo[2], lo[3]);
    }

    // TDM descriptor group 1 (constant): 2D tile 256 dwords x 64 rows,
    // stride 256 dwords, pad 4 dwords (16B) after every 256 dwords -> DP=520.
    tdm_v8i g1 = {0, 0, 0, 0, 0, 0, 0, 0};
    if constexpr (FROM_WS) {
        g1[0] = (2 << 16)       // data_size = 4B
              | (1 << 20)       // pad_enable
              | (7 << 22)       // pad_interval: every 256 dwords
              | (3 << 25);      // pad_amount: 4 dwords (16B)
        g1[1] = (256 << 16);    // tensor_dim0[15:0] @ bits[63:48]
        g1[2] = (64 << 16);     // tensor_dim1[15:0] @ bits[95:80]
        g1[3] = (256 << 16);    // tile_dim0 @ bits[127:112]
        g1[4] = 64;             // tile_dim1 @ bits[143:128]
        g1[5] = 256;            // tensor_dim0_stride (dwords)
    }
    const tdm_v4i gz  = {0, 0, 0, 0};
    const tdm_v8i gz8 = {0, 0, 0, 0, 0, 0, 0, 0};

    float bestv[2][8];
    int   besti[2][8];
    #pragma unroll
    for (int t = 0; t < 2; ++t)
        #pragma unroll
        for (int i = 0; i < 8; ++i) { bestv[t][i] = 3.4e38f; besti[t][i] = 0; }

    const unsigned short* xh0 = xhi + (rBase + r) * DP;
    const unsigned short* xh1 = xhi + (rBase + 16 + r) * DP;
    const unsigned short* xl0 = xlo + (rBase + r) * DP;
    const unsigned short* xl1 = xlo + (rBase + 16 + r) * DP;
    const unsigned short* ch0 = chi + (cgBase + r) * DP;
    const unsigned short* ch1 = chi + (cgBase + 16 + r) * DP;
    const unsigned short* cl0 = clo + (cgBase + r) * DP;
    const unsigned short* cl1 = clo + (cgBase + 16 + r) * DP;

    for (int chunk = 0; chunk < NCHUNK; ++chunk) {
        const int c0 = chunk * NC;
        __syncthreads();   // previous chunk fully consumed
        if constexpr (FROM_WS) {
            if (wave == 0) {
                // two TDM loads: hi and lo planes of this 64-code chunk
                #pragma unroll
                for (int p = 0; p < 2; ++p) {
                    const unsigned short* gsrc =
                        (p ? cblo : cbhi) + (size_t)c0 * DD;
                    unsigned short* ldst = p ? clo : chi;
                    uint64_t ga = (uint64_t)(uintptr_t)gsrc;
                    uint32_t la = (uint32_t)(uintptr_t)ldst;
                    tdm_v4u g0;
                    g0[0] = 1u;                                   // count=1, user mode
                    g0[1] = la;                                   // lds_addr
                    g0[2] = (uint32_t)ga;                         // global_addr[31:0]
                    g0[3] = (uint32_t)((ga >> 32) & 0x1FFFFFFu)   // global_addr[56:32]
                          | (2u << 30);                           // type = image
                    __builtin_amdgcn_tensor_load_to_lds(g0, g1, gz, gz, gz8, 0);
                }
                __builtin_amdgcn_s_wait_tensorcnt(0);
            }
        } else {
            for (int e = tid; e < NC * D4; e += TPB) {
                int code = e >> 7;
                int c    = e & (D4 - 1);
                float4 v = cb4[(size_t)(c0 + code) * D4 + c];
                float vv[4] = {v.x, v.y, v.z, v.w};
                unsigned short hi[4], lo[4];
                #pragma unroll
                for (int j = 0; j < 4; ++j) {
                    hi[j] = f32_to_bf16_rne(vv[j]);
                    lo[j] = f32_to_bf16_rne(vv[j] - bf16u_to_f32(hi[j]));
                }
                int o = code * DP + c * 4;
                *(ushort4*)(&chi[o]) = make_ushort4(hi[0], hi[1], hi[2], hi[3]);
                *(ushort4*)(&clo[o]) = make_ushort4(lo[0], lo[1], lo[2], lo[3]);
            }
        }
        __syncthreads();   // chunk staged

        v8f a00 = {0,0,0,0,0,0,0,0}, a01 = {0,0,0,0,0,0,0,0};
        v8f a10 = {0,0,0,0,0,0,0,0}, a11 = {0,0,0,0,0,0,0,0};
        for (int k0 = 0; k0 < DD; k0 += 32) {
            FragAB ah0, al0, ah1, al1, bh0, bl0, bh1, bl1;
            ah0.u[0] = *(const uint4*)(xh0 + k0 + 8 * h);
            ah0.u[1] = *(const uint4*)(xh0 + k0 + 16 + 8 * h);
            al0.u[0] = *(const uint4*)(xl0 + k0 + 8 * h);
            al0.u[1] = *(const uint4*)(xl0 + k0 + 16 + 8 * h);
            ah1.u[0] = *(const uint4*)(xh1 + k0 + 8 * h);
            ah1.u[1] = *(const uint4*)(xh1 + k0 + 16 + 8 * h);
            al1.u[0] = *(const uint4*)(xl1 + k0 + 8 * h);
            al1.u[1] = *(const uint4*)(xl1 + k0 + 16 + 8 * h);
            bh0.u[0] = *(const uint4*)(ch0 + k0 + 8 * h);
            bh0.u[1] = *(const uint4*)(ch0 + k0 + 16 + 8 * h);
            bl0.u[0] = *(const uint4*)(cl0 + k0 + 8 * h);
            bl0.u[1] = *(const uint4*)(cl0 + k0 + 16 + 8 * h);
            bh1.u[0] = *(const uint4*)(ch1 + k0 + 8 * h);
            bh1.u[1] = *(const uint4*)(ch1 + k0 + 16 + 8 * h);
            bl1.u[0] = *(const uint4*)(cl1 + k0 + 8 * h);
            bl1.u[1] = *(const uint4*)(cl1 + k0 + 16 + 8 * h);
            // 4 independent accumulator chains, 3 bf16x3 terms each
            a00 = __builtin_amdgcn_wmma_f32_16x16x32_bf16(false, ah0.v, false, bh0.v, (short)0, a00, false, false);
            a01 = __builtin_amdgcn_wmma_f32_16x16x32_bf16(false, ah0.v, false, bh1.v, (short)0, a01, false, false);
            a10 = __builtin_amdgcn_wmma_f32_16x16x32_bf16(false, ah1.v, false, bh0.v, (short)0, a10, false, false);
            a11 = __builtin_amdgcn_wmma_f32_16x16x32_bf16(false, ah1.v, false, bh1.v, (short)0, a11, false, false);
            a00 = __builtin_amdgcn_wmma_f32_16x16x32_bf16(false, ah0.v, false, bl0.v, (short)0, a00, false, false);
            a01 = __builtin_amdgcn_wmma_f32_16x16x32_bf16(false, ah0.v, false, bl1.v, (short)0, a01, false, false);
            a10 = __builtin_amdgcn_wmma_f32_16x16x32_bf16(false, ah1.v, false, bl0.v, (short)0, a10, false, false);
            a11 = __builtin_amdgcn_wmma_f32_16x16x32_bf16(false, ah1.v, false, bl1.v, (short)0, a11, false, false);
            a00 = __builtin_amdgcn_wmma_f32_16x16x32_bf16(false, al0.v, false, bh0.v, (short)0, a00, false, false);
            a01 = __builtin_amdgcn_wmma_f32_16x16x32_bf16(false, al0.v, false, bh1.v, (short)0, a01, false, false);
            a10 = __builtin_amdgcn_wmma_f32_16x16x32_bf16(false, al1.v, false, bh0.v, (short)0, a10, false, false);
            a11 = __builtin_amdgcn_wmma_f32_16x16x32_bf16(false, al1.v, false, bh1.v, (short)0, a11, false, false);
        }

        // scores: ||c||^2 - 2*dot ; update running argmin
        float cn0  = cnorm[c0 + cgBase + r];
        float cn1  = cnorm[c0 + cgBase + 16 + r];
        int   cd0  = c0 + cgBase + r;
        int   cd1  = c0 + cgBase + 16 + r;
        #pragma unroll
        for (int i = 0; i < 8; ++i) {
            float s;
            s = cn0 - 2.0f * a00[i]; if (s < bestv[0][i]) { bestv[0][i] = s; besti[0][i] = cd0; }
            s = cn1 - 2.0f * a01[i]; if (s < bestv[0][i]) { bestv[0][i] = s; besti[0][i] = cd1; }
            s = cn0 - 2.0f * a10[i]; if (s < bestv[1][i]) { bestv[1][i] = s; besti[1][i] = cd0; }
            s = cn1 - 2.0f * a11[i]; if (s < bestv[1][i]) { bestv[1][i] = s; besti[1][i] = cd1; }
        }
    }

    // in-wave butterfly argmin over the 16 lanes sharing each row set
    #pragma unroll
    for (int t = 0; t < 2; ++t) {
        #pragma unroll
        for (int i = 0; i < 8; ++i) {
            float v = bestv[t][i];
            int idx = besti[t][i];
            #pragma unroll
            for (int off = 1; off < 16; off <<= 1) {
                float ov = __shfl_xor(v, off, 16);
                int   oi = __shfl_xor(idx, off, 16);
                if (ov < v || (ov == v && oi < idx)) { v = ov; idx = oi; }
            }
            if (r == 0) {
                int row = rBase + t * 16 + 8 * h + i;
                svalm[wave >> 1][row] = v;
                sidxm[wave >> 1][row] = idx;
            }
        }
    }
    __syncthreads();

    // merge the two code-group halves per row
    if (tid < ROWS) {
        float v0 = svalm[0][tid], v1 = svalm[1][tid];
        int   i0 = sidxm[0][tid], i1 = sidxm[1][tid];
        rowidx[tid] = (v1 < v0 || (v1 == v0 && i1 < i0)) ? i1 : i0;
    }
    __syncthreads();

    // gather codebook rows (L2-resident) into both outputs
    float4* outq4  = (float4*)outq;
    float4* outst4 = (float4*)outst;
    for (int e = tid; e < ROWS * D4; e += TPB) {
        int row = e >> 7;
        int c   = e & (D4 - 1);
        int idx = rowidx[row];
        float4 v = cb4[(size_t)idx * D4 + c];
        size_t o = (size_t)(R0 + row) * D4 + c;
        outq4[o]  = v;
        outst4[o] = v;
    }
}

extern "C" void kernel_launch(void* const* d_in, const int* in_sizes, int n_in,
                              void* d_out, int out_size, void* d_ws, size_t ws_size,
                              hipStream_t stream) {
    const float* x  = (const float*)d_in[0];   // [N, D] f32
    const float* cb = (const float*)d_in[1];   // [K, D] f32
    float* outq  = (float*)d_out;
    float* outst = outq + (size_t)NN * DD;

    float* cnorm = (float*)d_ws;                                  // KK floats
    unsigned short* cbhi = (unsigned short*)((char*)d_ws + (size_t)KK * 4);
    unsigned short* cblo = cbhi + (size_t)KK * DD;
    size_t need = (size_t)KK * 4 + 2 * (size_t)KK * DD * 2;       // ~16.8 MB

    vq_cnorm_kernel<<<KK / 256, 256, 0, stream>>>(cb, cnorm);
    if (ws_size >= need) {
        vq_split_cb_kernel<<<(KK * DD / 4) / 256, 256, 0, stream>>>(cb, cbhi, cblo);
        vq_main_kernel<true><<<NN / ROWS, TPB, 0, stream>>>(x, cb, cbhi, cblo, cnorm, outq, outst);
    } else {
        vq_main_kernel<false><<<NN / ROWS, TPB, 0, stream>>>(x, cb, cbhi, cblo, cnorm, outq, outst);
    }
}
